// Attention_40681930227768
// MI455X (gfx1250) — compile-verified, matrix operand order checked by reading, not run on hardware
//
#include <hip/hip_runtime.h>

// ---------------- problem constants ----------------
#define B_    32
#define N_    1024
#define DIM_  512
#define H_    8
#define DH_   64
#define QKV3  (3 * H_ * DH_)          // 1536
#define MTOT  (B_ * N_)               // 32768
#define LDPAD 8
static constexpr float SCALE = 0.125f;  // DH^-0.5

// ---------------- types ----------------
typedef __attribute__((ext_vector_type(16))) __bf16         bf16x16;
typedef __attribute__((ext_vector_type(16))) unsigned short u16x16;
typedef __attribute__((ext_vector_type(8)))  float          f32x8;

// f32 -> bf16 bits, round-to-nearest-even
__device__ __forceinline__ unsigned short bf16b(float f) {
    unsigned int u = __float_as_uint(f);
    unsigned int r = u + 0x7FFFu + ((u >> 16) & 1u);
    return (unsigned short)(r >> 16);
}

// Build a 16x32 bf16 A/B fragment for v_wmma_f32_16x16x32_bf16.
// ISA 7.12.2 (16-bit A 16x32): lane half 0 holds K = {0..7, 16..23},
// lane half 1 holds K = {8..15, 24..31}; pairs packed per VGPR.
// `row` points at the start of this lane's 32-element K span (LDS or global).
__device__ __forceinline__ bf16x16 frag_ld(const unsigned short* row, int half) {
    union { u16x16 u; bf16x16 h; } f;
    const unsigned short* p0 = row + half * 8;
#pragma unroll
    for (int j = 0; j < 8; ++j) {
        f.u[j]     = p0[j];        // K = half*8 + j
        f.u[8 + j] = p0[16 + j];   // K = 16 + half*8 + j
    }
    return f.h;
}

__device__ __forceinline__ f32x8 wmma_bf16(bf16x16 a, bf16x16 b, f32x8 c) {
    return __builtin_amdgcn_wmma_f32_16x16x32_bf16(
        false, a, false, b, (short)0, c, false, false);
}

// =====================================================================
// Kernel 1: QKV projection.  C[32768 x 1536] = x @ W_qkv, bf16 out into
// q/k/v tensors laid out [B,H,N,DH]; SCALE folded into q.
// =====================================================================
__global__ __launch_bounds__(256) void qkv_gemm_kernel(
    const float* __restrict__ x, const float* __restrict__ Wqkv,
    unsigned short* __restrict__ q, unsigned short* __restrict__ k,
    unsigned short* __restrict__ v)
{
    __shared__ unsigned short As[128][32 + LDPAD];
    __shared__ unsigned short Bt[128][32 + LDPAD];   // B transposed: [n][k]

    const int tid = threadIdx.x;
    const int lane = tid & 31, w = tid >> 5;
    const int half = lane >> 4, m16 = lane & 15;
    const int wr = w >> 1, wc = w & 1;               // 4x2 wave grid
    const int bn = blockIdx.x % (QKV3 / 128);
    const int bm = blockIdx.x / (QKV3 / 128);
    const int row0 = bm * 128, col0 = bn * 128;

    f32x8 acc[2][4];
#pragma unroll
    for (int i = 0; i < 2; ++i)
#pragma unroll
        for (int j = 0; j < 4; ++j)
#pragma unroll
            for (int r = 0; r < 8; ++r) acc[i][j][r] = 0.0f;

    for (int k0 = 0; k0 < DIM_; k0 += 32) {
        __syncthreads();
        // stage A tile 128x32, f32 -> bf16
#pragma unroll
        for (int i = 0; i < 4; ++i) {
            int idx = tid + i * 256;                 // float4 units
            int r = idx >> 3, c4 = (idx & 7) * 4;
            const float4 f = *reinterpret_cast<const float4*>(
                &x[(size_t)(row0 + r) * DIM_ + k0 + c4]);
            ushort4 u;
            u.x = bf16b(f.x); u.y = bf16b(f.y); u.z = bf16b(f.z); u.w = bf16b(f.w);
            *reinterpret_cast<ushort4*>(&As[r][c4]) = u;
        }
        // stage B tile 32x128 transposed into Bt[n][k]
#pragma unroll
        for (int i = 0; i < 4; ++i) {
            int idx = tid + i * 256;
            int kk = idx >> 5, n4 = (idx & 31) * 4;
            const float4 f = *reinterpret_cast<const float4*>(
                &Wqkv[(size_t)(k0 + kk) * QKV3 + col0 + n4]);
            Bt[n4 + 0][kk] = bf16b(f.x);
            Bt[n4 + 1][kk] = bf16b(f.y);
            Bt[n4 + 2][kk] = bf16b(f.z);
            Bt[n4 + 3][kk] = bf16b(f.w);
        }
        __syncthreads();

        bf16x16 a0 = frag_ld(&As[wr * 32 + m16][0], half);
        bf16x16 a1 = frag_ld(&As[wr * 32 + 16 + m16][0], half);
#pragma unroll
        for (int j = 0; j < 4; ++j) {
            bf16x16 b = frag_ld(&Bt[wc * 64 + j * 16 + m16][0], half);
            acc[0][j] = wmma_bf16(a0, b, acc[0][j]);
            acc[1][j] = wmma_bf16(a1, b, acc[1][j]);
        }
    }

    // epilogue: scatter bf16 into q/k/v [B,H,N,DH]; scale q
#pragma unroll
    for (int ti = 0; ti < 2; ++ti)
#pragma unroll
        for (int tj = 0; tj < 4; ++tj) {
            int colg = col0 + wc * 64 + tj * 16 + m16;
            int which = colg / (H_ * DH_);
            int rr = colg % (H_ * DH_);
            int h = rr >> 6, d = rr & 63;
            unsigned short* dst = (which == 0) ? q : (which == 1) ? k : v;
            float mul = (which == 0) ? SCALE : 1.0f;
#pragma unroll
            for (int r = 0; r < 8; ++r) {
                int rowg = row0 + wr * 32 + ti * 16 + half * 8 + r;
                int bb = rowg >> 10, n = rowg & (N_ - 1);
                dst[(((size_t)bb * H_ + h) * N_ + n) * DH_ + d] =
                    bf16b(acc[ti][tj][r] * mul);
            }
        }
}

// =====================================================================
// Kernel 2: flash attention with post-softmax decay mask.
// grid = (B*H) * (N/128); 8 waves, each owns 16 query rows.
// =====================================================================
__global__ __launch_bounds__(256) void attn_kernel(
    const unsigned short* __restrict__ q,
    const unsigned short* __restrict__ kk,
    const unsigned short* __restrict__ vv,
    const float* __restrict__ mask,
    unsigned short* __restrict__ obuf)
{
    __shared__ unsigned short Ks[64][64 + LDPAD];        // [j][d]  (== K^T LDS form)
    __shared__ unsigned short Vt[64][64 + LDPAD];        // [d][j]  (V transposed)
    __shared__ unsigned short Ps[8][16][64 + LDPAD];     // per-wave P scratch

    const int tid = threadIdx.x;
    const int lane = tid & 31, w = tid >> 5;
    const int half = lane >> 4, m16 = lane & 15;
    const int bh = blockIdx.x >> 3;
    const int rowblk = blockIdx.x & 7;
    const int b = bh >> 3, h = bh & 7;
    const int row0 = rowblk * 128 + w * 16;              // this wave's rows

    // resident Q fragments (rows row0..row0+15, Kdim = DH = 64 -> 2 chunks)
    const unsigned short* qrow = q + ((size_t)bh * N_ + row0 + m16) * DH_;
    bf16x16 qa0 = frag_ld(qrow + 0, half);
    bf16x16 qa1 = frag_ld(qrow + 32, half);

    float mrow[8], lrow[8];
    f32x8 o[4];
#pragma unroll
    for (int r = 0; r < 8; ++r) { mrow[r] = -1e30f; lrow[r] = 0.0f; }
#pragma unroll
    for (int t = 0; t < 4; ++t)
#pragma unroll
        for (int r = 0; r < 8; ++r) o[t][r] = 0.0f;

    for (int jt = 0; jt < N_ / 64; ++jt) {
        const int j0 = jt * 64;
        __syncthreads();
        // stage K tile (row-major bf16 copy, b128 chunks)
#pragma unroll
        for (int i = 0; i < 2; ++i) {
            int chunk = tid + i * 256;                   // 512 chunks of 8 u16
            int j = chunk >> 3, c8 = (chunk & 7) * 8;
            *reinterpret_cast<uint4*>(&Ks[j][c8]) =
                *reinterpret_cast<const uint4*>(
                    &kk[((size_t)bh * N_ + j0 + j) * DH_ + c8]);
        }
        // stage V transposed
#pragma unroll
        for (int i = 0; i < 16; ++i) {
            int idx = tid + i * 256;
            int d = idx & 63, j = idx >> 6;
            Vt[d][j] = vv[((size_t)bh * N_ + j0 + j) * DH_ + d];
        }
        __syncthreads();

        // S = Q K^T : 4 column tiles of 16
        f32x8 s[4];
#pragma unroll
        for (int t = 0; t < 4; ++t) {
#pragma unroll
            for (int r = 0; r < 8; ++r) s[t][r] = 0.0f;
            bf16x16 kb0 = frag_ld(&Ks[t * 16 + m16][0], half);
            bf16x16 kb1 = frag_ld(&Ks[t * 16 + m16][32], half);
            s[t] = wmma_bf16(qa0, kb0, s[t]);
            s[t] = wmma_bf16(qa1, kb1, s[t]);
        }

        // online softmax (denominator uses UNMASKED exp; mask is post-softmax)
#pragma unroll
        for (int r = 0; r < 8; ++r) {
            float vmax = fmaxf(fmaxf(s[0][r], s[1][r]), fmaxf(s[2][r], s[3][r]));
            vmax = fmaxf(vmax, __shfl_xor(vmax, 1, 32));
            vmax = fmaxf(vmax, __shfl_xor(vmax, 2, 32));
            vmax = fmaxf(vmax, __shfl_xor(vmax, 4, 32));
            vmax = fmaxf(vmax, __shfl_xor(vmax, 8, 32));
            float mnew = fmaxf(mrow[r], vmax);
            float scl = __expf(mrow[r] - mnew);
            mrow[r] = mnew;
            lrow[r] *= scl;
#pragma unroll
            for (int t = 0; t < 4; ++t) o[t][r] *= scl;
            float rsum = 0.0f;
#pragma unroll
            for (int t = 0; t < 4; ++t) {
                float p = __expf(s[t][r] - mnew);
                s[t][r] = p;
                rsum += p;
            }
            rsum += __shfl_xor(rsum, 1, 32);
            rsum += __shfl_xor(rsum, 2, 32);
            rsum += __shfl_xor(rsum, 4, 32);
            rsum += __shfl_xor(rsum, 8, 32);
            lrow[r] += rsum;
        }

        // apply decay mask, park P in per-wave LDS (accum layout -> A layout)
        const int rg0 = rowblk * 128 + w * 16 + half * 8;
#pragma unroll
        for (int t = 0; t < 4; ++t) {
            int colg = j0 + t * 16 + m16;
#pragma unroll
            for (int r = 0; r < 8; ++r) {
                float pm = s[t][r] * mask[((size_t)h * N_ + (rg0 + r)) * N_ + colg];
                Ps[w][half * 8 + r][t * 16 + m16] = bf16b(pm);
            }
        }
        // same-wave DS ops are in-order: safe to read back without barrier
        bf16x16 pa0 = frag_ld(&Ps[w][m16][0], half);
        bf16x16 pa1 = frag_ld(&Ps[w][m16][32], half);

        // O += P @ V
#pragma unroll
        for (int t = 0; t < 4; ++t) {
            bf16x16 vb0 = frag_ld(&Vt[t * 16 + m16][0], half);
            bf16x16 vb1 = frag_ld(&Vt[t * 16 + m16][32], half);
            o[t] = wmma_bf16(pa0, vb0, o[t]);
            o[t] = wmma_bf16(pa1, vb1, o[t]);
        }
    }

    // finalize: divide by softmax denominator, write bf16 [B,N,H*DH]
#pragma unroll
    for (int r = 0; r < 8; ++r) {
        float inv = 1.0f / lrow[r];
#pragma unroll
        for (int t = 0; t < 4; ++t) o[t][r] *= inv;
    }
#pragma unroll
    for (int t = 0; t < 4; ++t)
#pragma unroll
        for (int r = 0; r < 8; ++r) {
            int n = rowblk * 128 + w * 16 + half * 8 + r;
            int col = h * DH_ + t * 16 + m16;
            obuf[((size_t)b * N_ + n) * (H_ * DH_) + col] = bf16b(o[t][r]);
        }
}

// =====================================================================
// Kernel 3: output projection.  out[32768 x 512] = obuf @ W_out + b_out
// =====================================================================
__global__ __launch_bounds__(256) void out_gemm_kernel(
    const unsigned short* __restrict__ A,   // bf16 [32768][512]
    const float* __restrict__ W,            // f32  [512][512]
    const float* __restrict__ bias,         // f32  [512]
    float* __restrict__ out)                // f32  [32768][512]
{
    __shared__ unsigned short As[128][32 + LDPAD];
    __shared__ unsigned short Bt[128][32 + LDPAD];

    const int tid = threadIdx.x;
    const int lane = tid & 31, w = tid >> 5;
    const int half = lane >> 4, m16 = lane & 15;
    const int wr = w >> 1, wc = w & 1;
    const int bn = blockIdx.x & 3;                    // 512/128
    const int bm = blockIdx.x >> 2;
    const int row0 = bm * 128, col0 = bn * 128;

    f32x8 acc[2][4];
#pragma unroll
    for (int i = 0; i < 2; ++i)
#pragma unroll
        for (int j = 0; j < 4; ++j)
#pragma unroll
            for (int r = 0; r < 8; ++r) acc[i][j][r] = 0.0f;

    for (int k0 = 0; k0 < DIM_; k0 += 32) {
        __syncthreads();
        // stage A (already bf16): b128 copies
#pragma unroll
        for (int i = 0; i < 2; ++i) {
            int chunk = tid + i * 256;                // 512 chunks of 8 u16
            int r = chunk >> 2, c8 = (chunk & 3) * 8;
            *reinterpret_cast<uint4*>(&As[r][c8]) =
                *reinterpret_cast<const uint4*>(
                    &A[(size_t)(row0 + r) * DIM_ + k0 + c8]);
        }
        // stage W transposed, f32 -> bf16
#pragma unroll
        for (int i = 0; i < 4; ++i) {
            int idx = tid + i * 256;
            int kk = idx >> 5, n4 = (idx & 31) * 4;
            const float4 f = *reinterpret_cast<const float4*>(
                &W[(size_t)(k0 + kk) * DIM_ + col0 + n4]);
            Bt[n4 + 0][kk] = bf16b(f.x);
            Bt[n4 + 1][kk] = bf16b(f.y);
            Bt[n4 + 2][kk] = bf16b(f.z);
            Bt[n4 + 3][kk] = bf16b(f.w);
        }
        __syncthreads();

        bf16x16 a0 = frag_ld(&As[wr * 32 + m16][0], half);
        bf16x16 a1 = frag_ld(&As[wr * 32 + 16 + m16][0], half);
#pragma unroll
        for (int j = 0; j < 4; ++j) {
            bf16x16 b = frag_ld(&Bt[wc * 64 + j * 16 + m16][0], half);
            acc[0][j] = wmma_bf16(a0, b, acc[0][j]);
            acc[1][j] = wmma_bf16(a1, b, acc[1][j]);
        }
    }

#pragma unroll
    for (int ti = 0; ti < 2; ++ti)
#pragma unroll
        for (int tj = 0; tj < 4; ++tj) {
            int colg = col0 + wc * 64 + tj * 16 + m16;
            float bb = bias[colg];
#pragma unroll
            for (int r = 0; r < 8; ++r) {
                int rowg = row0 + wr * 32 + ti * 16 + half * 8 + r;
                out[(size_t)rowg * DIM_ + colg] = acc[ti][tj][r] + bb;
            }
        }
}

// =====================================================================
extern "C" void kernel_launch(void* const* d_in, const int* in_sizes, int n_in,
                              void* d_out, int out_size, void* d_ws, size_t ws_size,
                              hipStream_t stream) {
    const float* x     = (const float*)d_in[0];
    const float* Wqkv  = (const float*)d_in[1];
    const float* Wout  = (const float*)d_in[2];
    const float* bout  = (const float*)d_in[3];
    const float* mask  = (const float*)d_in[4];
    float* out = (float*)d_out;

    const size_t per = (size_t)B_ * H_ * N_ * DH_;   // 16,777,216 bf16 elems
    unsigned short* q    = (unsigned short*)d_ws;
    unsigned short* k    = q + per;
    unsigned short* v    = k + per;
    unsigned short* obuf = v + per;                  // [B,N,H*DH] bf16

    qkv_gemm_kernel<<<(MTOT / 128) * (QKV3 / 128), 256, 0, stream>>>(
        x, Wqkv, q, k, v);
    attn_kernel<<<B_ * H_ * (N_ / 128), 256, 0, stream>>>(
        q, k, v, mask, obuf);
    out_gemm_kernel<<<(MTOT / 128) * (DIM_ / 128), 256, 0, stream>>>(
        obuf, Wout, bout, out);
}
